// EncoderBlock_45509473468796
// MI455X (gfx1250) — matrix operand order repaired and checked by reading, not compiled, not run on hardware
//
#include <hip/hip_runtime.h>

// MI455X / gfx1250: wave32, WMMA f32_16x16x32_f16, LDS-staged weights, f32 L2 atomics.
// Streaming paths (x_edge in, x_edge1 out, edge_index) are non-temporal so the
// 192MB L2 stays dedicated to the 25.6MB node buffer absorbing 1.6GB of atomics.
// Lockstep loop + __syncthreads keeps the scheduler from hoisting loop-invariant
// bias/gamma/beta vectors into registers (>256 VGPR blow-up in free-running form).
// D->B repack uses v_cvt_pk_rtz_f16_f32 + v_pk_max_num_f16 explicitly.

typedef __attribute__((ext_vector_type(16))) _Float16 v16h;
typedef __attribute__((ext_vector_type(2)))  _Float16 v2h;
typedef __attribute__((ext_vector_type(8)))  float    v8f;
typedef __attribute__((ext_vector_type(4)))  float    v4f;
typedef __attribute__((ext_vector_type(2)))  int      v2i;

#define HID   64
#define INDIM 32
#define WAVES 8
#define TPB   256

__device__ __forceinline__ v8f wmma_f16(v16h a, v16h b, v8f c) {
  // D = A(16x32 f16) * B(32x16 f16) + C(16x16 f32)
  return __builtin_amdgcn_wmma_f32_16x16x32_f16(false, a, false, b, (short)0, c, false, false);
}

// single v_cvt_pk_rtz_f16_f32
__device__ __forceinline__ unsigned pack2(float x, float y) {
  auto r = __builtin_amdgcn_cvt_pkrtz(x, y);
  return __builtin_bit_cast(unsigned, r);
}

// packed ReLU on two f16: v_pk_max_num_f16
__device__ __forceinline__ unsigned relu_pk(unsigned u) {
  v2h x = __builtin_bit_cast(v2h, u);
  v2h z = { (_Float16)0, (_Float16)0 };
  x = __builtin_elementwise_max(x, z);
  return __builtin_bit_cast(unsigned, x);
}

union U16h { unsigned u[8]; v16h v; };

// 8 consecutive f32 starting at feat (16*t + 8*h)
__device__ __forceinline__ v8f load8(const float* __restrict__ p, int t, int h) {
  const float* q = p + 16 * t + 8 * h;
  v4f a = *(const v4f*)q;
  v4f b = *(const v4f*)(q + 4);
  v8f r;
  r[0]=a[0]; r[1]=a[1]; r[2]=a[2]; r[3]=a[3];
  r[4]=b[0]; r[5]=b[1]; r[6]=b[2]; r[7]=b[3];
  return r;
}

// Convert two D tiles (feats [0..15],[16..31] of a 32-feat block, lanes=edges)
// into the next layer's B operand (K=32 feats x 16 edges).  Only needs a
// cross-half lane swap (shfl_xor 16); everything else is lane-local.
// ReLU applied on packed f16 pairs (cvt_rtz then max == max then cvt_rtz).
template <bool RELU>
__device__ __forceinline__ v16h makeB(v8f dlo, v8f dhi, int h) {
  unsigned p0[4], p1[4];
#pragma unroll
  for (int w = 0; w < 4; ++w) {
    p0[w] = pack2(dlo[2*w], dlo[2*w+1]);
    p1[w] = pack2(dhi[2*w], dhi[2*w+1]);
    if (RELU) { p0[w] = relu_pk(p0[w]); p1[w] = relu_pk(p1[w]); }
  }
  U16h r;
#pragma unroll
  for (int w = 0; w < 4; ++w) {
    unsigned q0 = __shfl_xor(p0[w], 16, 32);
    unsigned q1 = __shfl_xor(p1[w], 16, 32);
    r.u[w]     = h ? q1    : p0[w];
    r.u[4 + w] = h ? p1[w] : q0;
  }
  return r.v;
}

__global__ void __launch_bounds__(TPB) zero_kernel(float* __restrict__ p, long n4) {
  long i = (long)blockIdx.x * blockDim.x + threadIdx.x;
  long stride = (long)gridDim.x * blockDim.x;
  v4f z; z[0] = 0.f; z[1] = 0.f; z[2] = 0.f; z[3] = 0.f;
  for (; i < n4; i += stride) ((v4f*)p)[i] = z;
}

__global__ void __launch_bounds__(TPB) edge_mlp_kernel(
    const float* __restrict__ x_edge, const int* __restrict__ eidx,
    const float* __restrict__ W1, const float* __restrict__ b1,
    const float* __restrict__ W2, const float* __restrict__ b2,
    const float* __restrict__ W3, const float* __restrict__ b3,
    const float* __restrict__ W4, const float* __restrict__ b4,
    const float* __restrict__ gammap, const float* __restrict__ betap,
    float* __restrict__ x_node, float* __restrict__ x_edge1,
    int E, int numTiles) {
  // Weights in A-operand f16 layout: L1 = 4 operands, L2..L4 = 8 operands each.
  // One operand = 32 lanes x 16 f16 = 1024 B (512 halfwords).
  __shared__ _Float16 sW[14336] __attribute__((aligned(32)));        // 28 KB
  __shared__ float sStage[WAVES][1088] __attribute__((aligned(16))); // 16 edges x 68 f32 per wave

  const int tid  = threadIdx.x;
  const int lane = tid & 31;
  const int wv   = tid >> 5;
  const int h    = lane >> 4;   // lane half
  const int eL   = lane & 15;   // edge-in-tile (B/D column)

  // ---- one-time cooperative weight conversion (f32 global -> f16 LDS, A layout) ----
  {
    const float* Ws[4] = { W1, W2, W3, W4 };
    const int layOff[4] = { 0, 2048, 6144, 10240 }; // halfword offsets
#pragma unroll
    for (int l = 0; l < 4; ++l) {
      const int khn   = (l == 0) ? 1 : 2;           // K/32
      const int total = 4 * khn * 512;
      const float* W  = Ws[l];
      for (int u = tid; u < total; u += TPB) {
        const int op = u >> 9;
        const int ln = (u >> 4) & 31;
        const int i  = u & 15;
        const int t  = op / khn;
        const int kh = op - t * khn;
        const int f  = 16 * t + (ln & 15);                    // A row (out-feat)
        const int k  = 32 * kh + 16 * (ln >> 4) + i;          // contiguous K map: k = 16h + s
        sW[layOff[l] + op * 512 + ln * 16 + i] = (_Float16)W[k * 64 + f];
      }
    }
  }
  __syncthreads();

  // Lockstep grid-stride loop: all waves of a block iterate together so the
  // __syncthreads (which also orders each wave's LDS staging) is uniform.
  for (int base = blockIdx.x * WAVES; base < numTiles; base += gridDim.x * WAVES) {
    const int tile = base + wv;                 // wave-uniform
    const bool active = tile < numTiles;
    if (active) {
      const int edge  = tile * 16 + eL;
      const int edgeC = edge < E ? edge : (E - 1);

      // prefetch next iteration's x rows
      {
        const int tn = tile + gridDim.x * WAVES;
        if (tn < numTiles) {
          const int en = tn * 16 + eL;
          __builtin_prefetch(x_edge + (size_t)(en < E ? en : (E - 1)) * INDIM + 16 * h, 0, 1);
        }
      }

      // ---- B operand for layer 1: X^T (K=32 feats x 16 edges), f32 -> f16, NT ----
      const v4f* xp = (const v4f*)(x_edge + (size_t)edgeC * INDIM + 16 * h);
      v16h bx;
#pragma unroll
      for (int i = 0; i < 4; ++i) {
        v4f f4 = __builtin_nontemporal_load(xp + i);
        bx[4*i+0] = (_Float16)f4[0]; bx[4*i+1] = (_Float16)f4[1];
        bx[4*i+2] = (_Float16)f4[2]; bx[4*i+3] = (_Float16)f4[3];
      }

      v8f acc[4];
      // ---- layer 1: H^T = W1^T (64x32) * X^T (32x16) ----
#pragma unroll
      for (int t = 0; t < 4; ++t) {
        v8f c = load8(b1, t, h);
        v16h a = *(const v16h*)&sW[0 + t * 512 + lane * 16];
        acc[t] = wmma_f16(a, bx, c);
      }
      v16h blo = makeB<true>(acc[0], acc[1], h);
      v16h bhi = makeB<true>(acc[2], acc[3], h);

      // ---- layer 2 (K=64: two chained WMMAs) ----
#pragma unroll
      for (int t = 0; t < 4; ++t) {
        v8f c = load8(b2, t, h);
        c = wmma_f16(*(const v16h*)&sW[2048 + (2*t+0) * 512 + lane * 16], blo, c);
        c = wmma_f16(*(const v16h*)&sW[2048 + (2*t+1) * 512 + lane * 16], bhi, c);
        acc[t] = c;
      }
      blo = makeB<true>(acc[0], acc[1], h);
      bhi = makeB<true>(acc[2], acc[3], h);

      // ---- layer 3 ----
#pragma unroll
      for (int t = 0; t < 4; ++t) {
        v8f c = load8(b3, t, h);
        c = wmma_f16(*(const v16h*)&sW[6144 + (2*t+0) * 512 + lane * 16], blo, c);
        c = wmma_f16(*(const v16h*)&sW[6144 + (2*t+1) * 512 + lane * 16], bhi, c);
        acc[t] = c;
      }
      blo = makeB<true>(acc[0], acc[1], h);
      bhi = makeB<true>(acc[2], acc[3], h);

      // ---- layer 4 (no ReLU) ----
#pragma unroll
      for (int t = 0; t < 4; ++t) {
        v8f c = load8(b4, t, h);
        c = wmma_f16(*(const v16h*)&sW[10240 + (2*t+0) * 512 + lane * 16], blo, c);
        c = wmma_f16(*(const v16h*)&sW[10240 + (2*t+1) * 512 + lane * 16], bhi, c);
        acc[t] = c;
      }

      // ---- LayerNorm over 64 feats: each lane holds 32 feats of its edge,
      //      partner lane (xor 16) holds the other 32 ----
      float s1 = 0.f, s2 = 0.f;
#pragma unroll
      for (int t = 0; t < 4; ++t)
#pragma unroll
        for (int v = 0; v < 8; ++v) { float x = acc[t][v]; s1 += x; s2 += x * x; }
      s1 += __shfl_xor(s1, 16, 32);
      s2 += __shfl_xor(s2, 16, 32);
      const float mu  = s1 * (1.f / 64.f);
      const float var = s2 * (1.f / 64.f) - mu * mu;
      const float rs  = rsqrtf(var + 1e-5f);
#pragma unroll
      for (int t = 0; t < 4; ++t) {
        v8f g  = load8(gammap, t, h);
        v8f bb = load8(betap,  t, h);
#pragma unroll
        for (int v = 0; v < 8; ++v)
          acc[t][v] = (acc[t][v] - mu) * rs * g[v] + bb[v];
      }

      // ---- scatter-add both endpoints (L2-resident node buffer) ----
      if (edge < E) {
        const v2i nn = __builtin_nontemporal_load((const v2i*)(eidx + 2 * edge));
        float* r0 = x_node + (size_t)nn.x * HID + 8 * h;
        float* r1 = x_node + (size_t)nn.y * HID + 8 * h;
#pragma unroll
        for (int t = 0; t < 4; ++t)
#pragma unroll
          for (int v = 0; v < 8; ++v) {
            unsafeAtomicAdd(r0 + 16 * t + v, acc[t][v]);
            unsafeAtomicAdd(r1 + 16 * t + v, acc[t][v]);
          }
      }

      // ---- stage x_edge1 to LDS (padded rows) for coalesced b128 stores ----
#pragma unroll
      for (int t = 0; t < 4; ++t) {
        float* sp = &sStage[wv][eL * 68 + 16 * t + 8 * h];
        v4f lo; lo[0] = acc[t][0]; lo[1] = acc[t][1]; lo[2] = acc[t][2]; lo[3] = acc[t][3];
        v4f hi; hi[0] = acc[t][4]; hi[1] = acc[t][5]; hi[2] = acc[t][6]; hi[3] = acc[t][7];
        *(v4f*)sp       = lo;
        *(v4f*)(sp + 4) = hi;
      }
    }
    __syncthreads();   // block-uniform; orders wave-local LDS write->read
    if (active) {
#pragma unroll
      for (int j = 0; j < 8; ++j) {
        const int q = j * 32 + lane;
        const int e = q >> 4, c4 = q & 15;
        v4f val = *(const v4f*)&sStage[wv][e * 68 + 4 * c4];
        const int edge2 = tile * 16 + e;
        if (edge2 < E)   // 512B/instr coalesced, non-temporal (write-once stream)
          __builtin_nontemporal_store(val, (v4f*)(x_edge1 + (size_t)edge2 * HID + 4 * c4));
      }
    }
    __syncthreads();   // staging reuse guard
  }
}

extern "C" void kernel_launch(void* const* d_in, const int* in_sizes, int n_in,
                              void* d_out, int out_size, void* d_ws, size_t ws_size,
                              hipStream_t stream) {
  const float* x_edge = (const float*)d_in[0];
  const int*   eidx   = (const int*)d_in[1];
  const float* W1 = (const float*)d_in[3];  const float* b1 = (const float*)d_in[4];
  const float* W2 = (const float*)d_in[5];  const float* b2 = (const float*)d_in[6];
  const float* W3 = (const float*)d_in[7];  const float* b3 = (const float*)d_in[8];
  const float* W4 = (const float*)d_in[9];  const float* b4 = (const float*)d_in[10];
  const float* gm = (const float*)d_in[11]; const float* bt = (const float*)d_in[12];

  const long E = (long)in_sizes[0] / INDIM;
  const long nodeFloats = (long)out_size - E * HID;   // num_nodes * 64
  float* x_node  = (float*)d_out;
  float* x_edge1 = x_node + nodeFloats;

  // zero the node accumulator (outputs are poisoned before timing)
  {
    long n4 = nodeFloats / 4;
    long zb = (n4 + TPB - 1) / TPB; if (zb > 4096) zb = 4096;
    zero_kernel<<<(int)zb, TPB, 0, stream>>>(x_node, n4);
  }

  const long numTiles = (E + 15) / 16;
  long blocks = (numTiles + WAVES - 1) / WAVES; if (blocks > 4096) blocks = 4096;
  edge_mlp_kernel<<<(int)blocks, TPB, 0, stream>>>(
      x_edge, eidx, W1, b1, W2, b2, W3, b3, W4, b4, gm, bt,
      x_node, x_edge1, (int)E, (int)numTiles);
}
